// BitNetLinear_43714177139353
// MI455X (gfx1250) — 8-bit matrix operand order repaired and checked by reading, compiled, not run on hardware
//
#include <hip/hip_runtime.h>
#include <hip/hip_bf16.h>

typedef __attribute__((ext_vector_type(8))) int   v8i;
typedef __attribute__((ext_vector_type(4))) int   v4i;
typedef __attribute__((ext_vector_type(2))) int   v2i;
typedef __attribute__((ext_vector_type(4))) float v4f;

#define IN_F   1024
#define OUT_F  1024
#define M_TOT  (8 * 4096)

#define BM 128
#define BN 128
#define BK 64

#if defined(__has_builtin)
#if __has_builtin(__builtin_amdgcn_global_load_async_to_lds_b128)
#define HAS_ASYNC_LDS 1
#endif
#endif
#ifndef HAS_ASYNC_LDS
#define HAS_ASYNC_LDS 0
#endif

typedef __attribute__((address_space(1))) v4i as1_v4i;
typedef __attribute__((address_space(3))) v4i as3_v4i;

__device__ __forceinline__ int clamp_i8(int v) {
    v = v < -128 ? -128 : v;
    v = v >  127 ?  127 : v;
    return v;
}

__device__ __forceinline__ int pack4_quant(v4f f, float s) {
    int a = clamp_i8((int)rintf(f.x * s));
    int b = clamp_i8((int)rintf(f.y * s));
    int c = clamp_i8((int)rintf(f.z * s));
    int d = clamp_i8((int)rintf(f.w * s));
    return (a & 255) | ((b & 255) << 8) | ((c & 255) << 16) | ((d & 255) << 24);
}

__device__ __forceinline__ void copy_b128_to_lds(const signed char* g, signed char* l) {
#if HAS_ASYNC_LDS
    __builtin_amdgcn_global_load_async_to_lds_b128(
        (as1_v4i*)g, (as3_v4i*)l, 0, 0);
#else
    *(v4i*)l = *(const v4i*)g;
#endif
}

__device__ __forceinline__ void wait_async_zero() {
#if HAS_ASYNC_LDS
#if defined(__has_builtin) && __has_builtin(__builtin_amdgcn_s_wait_asynccnt)
    __builtin_amdgcn_s_wait_asynccnt(0);
#else
    asm volatile("s_wait_asynccnt 0" ::: "memory");
#endif
#endif
}

__global__ void init_ws_kernel(float* wsf, unsigned* wsu) {
    if (threadIdx.x == 0) { wsf[0] = 0.0f; wsu[1] = 0u; }
}

__global__ void reduce_wsum_kernel(const float* __restrict__ w, float* wsf) {
    __shared__ float red[256];
    const v4f* w4 = (const v4f*)w;
    const int  n4 = (OUT_F * IN_F) / 4;
    float s = 0.0f;
    for (int i = blockIdx.x * blockDim.x + threadIdx.x; i < n4;
         i += gridDim.x * blockDim.x) {
        v4f v = w4[i];
        s += fabsf(v.x) + fabsf(v.y) + fabsf(v.z) + fabsf(v.w);
    }
    red[threadIdx.x] = s;
    __syncthreads();
    for (int off = 128; off > 0; off >>= 1) {
        if (threadIdx.x < off) red[threadIdx.x] += red[threadIdx.x + off];
        __syncthreads();
    }
    if (threadIdx.x == 0) atomicAdd(&wsf[0], red[0]);
}

__global__ void reduce_xmax_kernel(const float* __restrict__ x, unsigned* wsu, int n4) {
    __shared__ float red[256];
    const v4f* x4 = (const v4f*)x;
    float m = 0.0f;
    for (int i = blockIdx.x * blockDim.x + threadIdx.x; i < n4;
         i += gridDim.x * blockDim.x) {
        v4f v = x4[i];
        m = fmaxf(m, fmaxf(fmaxf(fabsf(v.x), fabsf(v.y)),
                           fmaxf(fabsf(v.z), fabsf(v.w))));
    }
    red[threadIdx.x] = m;
    __syncthreads();
    for (int off = 128; off > 0; off >>= 1) {
        if (threadIdx.x < off)
            red[threadIdx.x] = fmaxf(red[threadIdx.x], red[threadIdx.x + off]);
        __syncthreads();
    }
    if (threadIdx.x == 0) atomicMax(&wsu[1], __float_as_uint(red[0]));
}

__global__ void quant_weight_kernel(const float* __restrict__ w,
                                    const float* __restrict__ wsf,
                                    int* __restrict__ qw) {
    const int i = blockIdx.x * blockDim.x + threadIdx.x;
    const float wsc = wsf[0] * (1.0f / (float)(OUT_F * IN_F));
    const float thr = 0.5f * wsc;
    v4f v = ((const v4f*)w)[i];
    int q0 = (fabsf(v.x) > thr) ? (v.x > 0.0f ? 1 : -1) : 0;
    int q1 = (fabsf(v.y) > thr) ? (v.y > 0.0f ? 1 : -1) : 0;
    int q2 = (fabsf(v.z) > thr) ? (v.z > 0.0f ? 1 : -1) : 0;
    int q3 = (fabsf(v.w) > thr) ? (v.w > 0.0f ? 1 : -1) : 0;
    qw[i] = (q0 & 255) | ((q1 & 255) << 8) | ((q2 & 255) << 16) | ((q3 & 255) << 24);
}

__global__ void quant_x_kernel(const float* __restrict__ x,
                               const unsigned* __restrict__ wsu,
                               int* __restrict__ qx) {
    const float imax  = __uint_as_float(wsu[1]);
    const float inv_i = (imax > 0.0f) ? (127.0f / imax) : 0.0f;
    const int i = blockIdx.x * blockDim.x + threadIdx.x;
    const v4f* s = (const v4f*)x + (size_t)i * 2;
    v4f a = s[0];
    v4f b = s[1];
    v2i r;
    r.x = pack4_quant(a, inv_i);
    r.y = pack4_quant(b, inv_i);
    ((v2i*)qx)[i] = r;
}

__global__ __launch_bounds__(256, 1) void bitnet_gemm_kernel(
    const signed char* __restrict__ qx,
    const signed char* __restrict__ qw,
    const float* __restrict__ bias,
    const float* __restrict__ wsf,
    const unsigned* __restrict__ wsu,
    float* __restrict__ out)
{
    __shared__ __align__(16) signed char lds_a[2][BM * BK];
    __shared__ __align__(16) signed char lds_b[2][BN * BK];

    const int tid  = threadIdx.x;
    const int lane = tid & 31;
    const int wave = tid >> 5;
    const int wm   = wave & 3;
    const int wn   = wave >> 2;
    const int lo   = lane & 15;
    const int hi   = lane >> 4;

    const int m0 = blockIdx.y * BM;
    const int n0 = blockIdx.x * BN;

    const float wsc   = wsf[0] * (1.0f / (float)(OUT_F * IN_F));
    const float imax  = __uint_as_float(wsu[1]);
    const float i_sc  = imax * (1.0f / 127.0f);
    const float scale = wsc * i_sc;

    v8i acc[2][4];
#pragma unroll
    for (int tm = 0; tm < 2; ++tm)
#pragma unroll
        for (int tn = 0; tn < 4; ++tn)
            acc[tm][tn] = (v8i){0, 0, 0, 0, 0, 0, 0, 0};

    const int arow  = tid >> 1;
    const int ahalf = tid & 1;
    const signed char* gA = qx + (size_t)(m0 + arow) * IN_F + ahalf * 32;
    const signed char* gB = qw + (size_t)(n0 + arow) * IN_F + ahalf * 32;
    const int ldst = arow * BK + ahalf * 32;

    copy_b128_to_lds(gA +  0, &lds_a[0][ldst]);
    copy_b128_to_lds(gA + 16, &lds_a[0][ldst + 16]);
    copy_b128_to_lds(gB +  0, &lds_b[0][ldst]);
    copy_b128_to_lds(gB + 16, &lds_b[0][ldst + 16]);

    int p = 0;
    for (int k0 = 0; k0 < IN_F; k0 += BK) {
        wait_async_zero();
        __syncthreads();

        if (k0 + BK < IN_F) {
            const int kn = k0 + BK;
            copy_b128_to_lds(gA + kn,      &lds_a[p ^ 1][ldst]);
            copy_b128_to_lds(gA + kn + 16, &lds_a[p ^ 1][ldst + 16]);
            copy_b128_to_lds(gB + kn,      &lds_b[p ^ 1][ldst]);
            copy_b128_to_lds(gB + kn + 16, &lds_b[p ^ 1][ldst + 16]);
        }

        v8i afrag[2];
#pragma unroll
        for (int tm = 0; tm < 2; ++tm) {
            const int row = wm * 32 + tm * 16 + lo;
            const v2i* ab = (const v2i*)&lds_a[p][0];
            v8i a;
#pragma unroll
            for (int c = 0; c < 4; ++c) {
                v2i t = ab[row * 8 + (c >> 1) * 4 + hi * 2 + (c & 1)];
                a[2 * c]     = t.x;
                a[2 * c + 1] = t.y;
            }
            afrag[tm] = a;
        }
        v8i bfrag[4];
#pragma unroll
        for (int tn = 0; tn < 4; ++tn) {
            const int n = wn * 64 + tn * 16 + lo;
            const v4i* bb = (const v4i*)&lds_b[p][0];
            v4i t0 = bb[n * 4 + hi];
            v4i t1 = bb[n * 4 + hi + 2];
            v8i b;
            b[0] = t0.x; b[1] = t0.y; b[2] = t0.z; b[3] = t0.w;
            b[4] = t1.x; b[5] = t1.y; b[6] = t1.z; b[7] = t1.w;
            bfrag[tn] = b;
        }

#pragma unroll
        for (int tm = 0; tm < 2; ++tm)
#pragma unroll
            for (int tn = 0; tn < 4; ++tn)
                acc[tm][tn] = __builtin_amdgcn_wmma_i32_16x16x64_iu8(
                    true, afrag[tm], true, bfrag[tn], acc[tm][tn], false, false);

        __syncthreads();
        p ^= 1;
    }

    float bv[4];
#pragma unroll
    for (int tn = 0; tn < 4; ++tn)
        bv[tn] = bias[n0 + wn * 64 + tn * 16 + lo];

#pragma unroll
    for (int tm = 0; tm < 2; ++tm) {
        const int rbase = m0 + wm * 32 + tm * 16 + hi * 8;
#pragma unroll
        for (int tn = 0; tn < 4; ++tn) {
            const int col = n0 + wn * 64 + tn * 16 + lo;
#pragma unroll
            for (int r = 0; r < 8; ++r) {
                out[(size_t)(rbase + r) * OUT_F + col] =
                    (float)acc[tm][tn][r] * scale + bv[tn];
            }
        }
    }
}

extern "C" void kernel_launch(void* const* d_in, const int* in_sizes, int n_in,
                              void* d_out, int out_size, void* d_ws, size_t ws_size,
                              hipStream_t stream) {
    const float* x    = (const float*)d_in[0];
    const float* w    = (const float*)d_in[1];
    const float* bias = (const float*)d_in[2];
    float* out        = (float*)d_out;

    float*       wsf = (float*)d_ws;
    unsigned*    wsu = (unsigned*)d_ws;
    signed char* qw  = (signed char*)d_ws + 256;
    signed char* qx  = (signed char*)d_ws + 256 + (1 << 20);

    init_ws_kernel<<<1, 32, 0, stream>>>(wsf, wsu);

    reduce_wsum_kernel<<<512, 256, 0, stream>>>(w, wsf);

    const int n4x = (M_TOT * IN_F) / 4;
    reduce_xmax_kernel<<<2048, 256, 0, stream>>>(x, wsu, n4x);

    quant_weight_kernel<<<(OUT_F * IN_F / 4) / 256, 256, 0, stream>>>(w, wsf, (int*)qw);

    quant_x_kernel<<<(M_TOT * IN_F / 8) / 256, 256, 0, stream>>>(x, wsu, (int*)qx);

    dim3 grid(OUT_F / BN, M_TOT / BM);
    bitnet_gemm_kernel<<<grid, 256, 0, stream>>>(qx, qw, bias, wsf, wsu, out);
}
